// LocalBitPredictor_31018253811902
// MI455X (gfx1250) — matrix operand order repaired and checked by reading, not compiled, not run on hardware
//
#include <hip/hip_runtime.h>

typedef __attribute__((ext_vector_type(16))) _Float16 v16h;
typedef __attribute__((ext_vector_type(8)))  _Float16 v8h;
typedef __attribute__((ext_vector_type(8)))  float    v8f;

#define NB      8      // bit channels
#define HID     32     // hidden units
#define HW      256    // spatial H == W
#define TILE_Y  8      // rows per workgroup
#define NWAVES  8      // waves per workgroup (256 threads, wave32)
#define XSTRIP  128    // pixels in x per workgroup (NWAVES * 16)
#define ROWW    130    // staged row width: XSTRIP + 2 halo
#define SLOTH   (ROWW * NB)   // halves per ring-buffer slot (1040; 16B-aligned*2)

__global__ __launch_bounds__(256)
void lbp_wmma_kernel(const float* __restrict__ z,
                     const float* __restrict__ W1,
                     const float* __restrict__ b1,
                     const float* __restrict__ W2,
                     const float* __restrict__ b2,
                     float* __restrict__ out) {
    // 3-row ring buffer of z (f16), channel-innermost for b128 gathers.
    __shared__ __align__(16) _Float16 zs[3 * SLOTH];
    // Per-wave h transpose buffer: [pixel][hidden].
    __shared__ __align__(16) _Float16 hbuf[NWAVES][16][HID];

    const int tid  = threadIdx.x;
    const int lane = tid & 31;
    const int wave = tid >> 5;
    const int lg   = lane >> 4;   // lane group (0/1)
    const int m    = lane & 15;   // A-row / D-column index

    const int strip  = blockIdx.x;            // 0..1
    const int ystart = blockIdx.y * TILE_Y;   // 0..248
    const int bz     = blockIdx.z;            // batch
    const int xbase  = strip * XSTRIP;

    const float* zb = z   + (size_t)bz * NB * HW * HW;
    float*       ob = out + (size_t)bz * NB * HW * HW;

    // ---- Layer-1 weights as WMMA B operands (lane&15 = N col, halves = K in table order).
    // Features ordered K = q*8 + c (q = non-center tap, c = channel);
    // original W1 row f = c*9 + p; f == 71 is the dropped feature -> zero.
    v16h Bw[2][2];
    {
        const int n  = lane & 15;
        const int hh = lane >> 4;
#pragma unroll
        for (int ck = 0; ck < 2; ++ck) {
#pragma unroll
            for (int nt = 0; nt < 2; ++nt) {
                v16h bv;
#pragma unroll
                for (int j = 0; j < 16; ++j) {
                    int k = (j < 8) ? 8 * hh + j : 16 + 8 * hh + (j - 8);
                    int q = ck * 4 + (k >> 3);
                    int p = q < 4 ? q : q + 1;
                    int f = (k & 7) * 9 + p;
                    bv[j] = (f < 71) ? (_Float16)W1[f * HID + nt * 16 + n]
                                     : (_Float16)0.f;
                }
                Bw[ck][nt] = bv;
            }
        }
    }
    // ---- Layer-2 weights, N padded 8 -> 16 with zeros.
    v16h B2w;
#pragma unroll
    for (int j = 0; j < 16; ++j) {
        int k2 = (j < 8) ? 8 * (lane >> 4) + j : 16 + 8 * (lane >> 4) + (j - 8);
        B2w[j] = ((lane & 15) < 8) ? (_Float16)W2[k2 * NB + (lane & 15)] : (_Float16)0.f;
    }

    // ---- Bias-initialized accumulators (C layout: col N = lane&15).
    v8f cb1a, cb1b, cb2;
    {
        float t0 = b1[m];
        float t1 = b1[16 + m];
        float t2 = (m < 8) ? b2[m] : 0.f;
#pragma unroll
        for (int r = 0; r < 8; ++r) { cb1a[r] = t0; cb1b[r] = t1; cb2[r] = t2; }
    }

    // ---- Ring-buffer slot offsets (halves), rotated each row: no runtime mod.
    int s0  = ystart % 3;
    int sm1 = (s0 == 0) ? 2 : s0 - 1;
    int sp1 = (s0 == 2) ? 0 : s0 + 1;
    int om1 = sm1 * SLOTH;   // slot holding row y-1
    int o0  = s0  * SLOTH;   // slot holding row y
    int op1 = sp1 * SLOTH;   // slot holding row y+1

    // ---- Staging: threads 0..129 each own one x column, all 8 channels.
    // 8 contiguous f16 writes -> one ds_store_b128; prefetch next row (free).
    auto stage = [&](int gy, int soff) {
        if (tid < ROWW) {
            int  gx   = xbase - 1 + tid;
            bool xok  = (unsigned)gx < (unsigned)HW;
            bool ok   = xok && ((unsigned)gy < (unsigned)HW);
            const float* src  = zb + (gy << 8) + gx;
            const float* srcn = zb + ((gy + 1) << 8) + gx;
            bool pf = xok && ((unsigned)(gy + 1) < (unsigned)HW);
            _Float16* dst = zs + soff + tid * NB;
            _Float16 tmp[NB];
#pragma unroll
            for (int c = 0; c < NB; ++c) {
                tmp[c] = (_Float16)(ok ? src[c << 16] : 0.f);
                if (pf) __builtin_prefetch(&srcn[c << 16], 0, 1);
            }
#pragma unroll
            for (int c = 0; c < NB; ++c) dst[c] = tmp[c];
        }
    };

    stage(ystart - 1, om1);
    stage(ystart, o0);

    // ---- Loop-invariant per-lane gather x-offsets (halves) for the four
    // 8-half A groups. Taps (q -> p, di, dj):
    //  ck0 lo: p=lg       (row y-1, dj=lg)
    //  ck0 hi: p=2+lg     (row y-1 / y, dj = 2 / 0)
    //  ck1 lo: p=5+lg     (row y / y+1, dj = 2 / 0)
    //  ck1 hi: p=7+lg     (row y+1, dj = 1+lg)
    const int localx = wave * 16 + m;
    const int xo_a0 = (localx + lg) * NB;
    const int xo_b0 = (localx + (lg ? 0 : 2)) * NB;
    const int xo_a1 = (localx + (lg ? 0 : 2)) * NB;
    const int xo_b1 = (localx + 1 + lg) * NB;

    // ---- Loop-invariant output base: x = xbase + wave*16 + lg*8 + r (contig).
    float* obase = ob + ((size_t)m * HW * HW) + xbase + wave * 16 + lg * 8;

    for (int y = ystart; y < ystart + TILE_Y; ++y) {
        __syncthreads();            // previous compute done before restage
        stage(y + 1, op1);
        __syncthreads();            // staged rows visible before gather

        // ---- Gather A operands: 4 x ds_load_b128, addresses = cheap adds.
        int r_b0 = lg ? o0  : om1;
        int r_a1 = lg ? op1 : o0;
        v8h a0lo = *(const v8h*)(zs + om1 + xo_a0);
        v8h a0hi = *(const v8h*)(zs + r_b0 + xo_b0);
        v8h a1lo = *(const v8h*)(zs + r_a1 + xo_a1);
        v8h a1hi = *(const v8h*)(zs + op1 + xo_b1);
        v16h a0 = __builtin_shufflevector(a0lo, a0hi, 0, 1, 2, 3, 4, 5, 6, 7,
                                                      8, 9, 10, 11, 12, 13, 14, 15);
        v16h a1 = __builtin_shufflevector(a1lo, a1hi, 0, 1, 2, 3, 4, 5, 6, 7,
                                                      8, 9, 10, 11, 12, 13, 14, 15);

        // ---- Layer 1: h = relu(feat @ W1 + b1), K = 64 over 2 WMMA chunks.
        v8f h0 = __builtin_amdgcn_wmma_f32_16x16x32_f16(false, a0, false, Bw[0][0],
                                                        (short)0, cb1a, false, false);
        h0 = __builtin_amdgcn_wmma_f32_16x16x32_f16(false, a1, false, Bw[1][0],
                                                    (short)0, h0, false, false);
        v8f h1 = __builtin_amdgcn_wmma_f32_16x16x32_f16(false, a0, false, Bw[0][1],
                                                        (short)0, cb1b, false, false);
        h1 = __builtin_amdgcn_wmma_f32_16x16x32_f16(false, a1, false, Bw[1][1],
                                                    (short)0, h1, false, false);

        // ---- ReLU + transpose D(M=pixel,N=hidden) -> A2(lane=pixel, halves=K)
        // via per-wave LDS buffer (DS ops are in-order within a wave).
        _Float16* hb = &hbuf[wave][0][0];
#pragma unroll
        for (int r = 0; r < 8; ++r) {
            int pix = r + lg * 8;
            hb[pix * HID + m]      = (_Float16)fmaxf(h0[r], 0.f);
            hb[pix * HID + 16 + m] = (_Float16)fmaxf(h1[r], 0.f);
        }
        v8h lo = *(const v8h*)(hb + m * HID + lg * 8);
        v8h hi = *(const v8h*)(hb + m * HID + 16 + lg * 8);
        v16h a2 = __builtin_shufflevector(lo, hi, 0, 1, 2, 3, 4, 5, 6, 7,
                                                  8, 9, 10, 11, 12, 13, 14, 15);

        // ---- Layer 2: logits = h @ W2 + b2 (one WMMA, N padded to 16).
        v8f o = __builtin_amdgcn_wmma_f32_16x16x32_f16(false, a2, false, B2w,
                                                       (short)0, cb2, false, false);

        // ---- Store: lane col N = out channel (0..7 real), pixels contiguous
        // in x -> one 32B-aligned v8f store (2 x global_store_b128).
        if (m < 8)
            *(v8f*)(obase + (y << 8)) = o;

        // ---- Rotate ring-buffer slots.
        int t = om1; om1 = o0; o0 = op1; op1 = t;
    }
}

extern "C" void kernel_launch(void* const* d_in, const int* in_sizes, int n_in,
                              void* d_out, int out_size, void* d_ws, size_t ws_size,
                              hipStream_t stream) {
    (void)in_sizes; (void)n_in; (void)out_size; (void)d_ws; (void)ws_size;
    const float* z  = (const float*)d_in[0];
    const float* W1 = (const float*)d_in[1];
    const float* b1 = (const float*)d_in[2];
    const float* W2 = (const float*)d_in[3];
    const float* b2 = (const float*)d_in[4];
    float* out = (float*)d_out;

    dim3 grid(HW / XSTRIP, HW / TILE_Y, 32);   // (2, 32, 32)
    dim3 block(256);
    lbp_wmma_kernel<<<grid, block, 0, stream>>>(z, W1, b1, W2, b2, out);
}
